// DiehlCookNetwork_37915971289210
// MI455X (gfx1250) — compile-verified, matrix operand order checked by reading, not run on hardware
//
#include <hip/hip_runtime.h>
#include <hip/hip_bf16.h>

// ---------------------------------------------------------------------------
// Diehl&Cook SNN on MI455X (gfx1250, wave32, WMMA).
//   Kernel 1: tiled transpose (b,n,t)f32 -> (t,b*n)f16            (bandwidth)
//   Kernel 2: weight prep: relu(W_fe)^T f16, (relu(W_ie)*mask)^T f16 padded
//   Kernel 3: FF GEMM: 32x80 output tile per wave, v_wmma_f32_16x16x32_f16
//   Kernel 4: persistent scan: 16 WGs x 16 batch rows, recurrent WMMA + LIF
// ---------------------------------------------------------------------------

typedef __attribute__((ext_vector_type(16))) _Float16 v16h;
typedef __attribute__((ext_vector_type(8)))  _Float16 v8h;
typedef __attribute__((ext_vector_type(8)))  float    v8f;

#define N_INPUT 784
#define N_EXC   400
#define N_INH   400
#define BATCH   256
#define TSTEPS  350
#define K_INH_PAD 416              // 400 padded to 13*32
#define ROWS    (BATCH * N_INPUT)  // 200704

#define EXC_VREST  (-65.0f)
#define EXC_VRESET (-60.0f)
#define EXC_VTH    (-52.0f)
#define EXC_DECAY  (0.9900498337491681f)   // exp(-1/100)
#define INH_VREST  (-60.0f)
#define INH_VRESET (-45.0f)
#define INH_VTH    (-40.0f)
#define INH_DECAY  (0.9048374180359595f)   // exp(-1/10)
#define WEIGHT_EI  (50.0f)

// ---- WMMA fragment helpers (16-bit A/B layout per CDNA5 ISA 7.12.2) -------
// A (16x32, row-major source, leading dim ld):
//   lane<16 : row M=lane,    K = {0..7} (v0..3) and {16..23} (v4..7)
//   lane>=16: row M=lane-16, K = {8..15}        and {24..31}
// B (32x16 KxN) is loaded from an N-major (transposed) buffer so each lane's
// column is a contiguous row: identical access pattern as A.
template <bool TAIL16>
__device__ __forceinline__ v16h load_frag(const _Float16* p, int ld) {
    unsigned lane = threadIdx.x & 31u;
    unsigned m    = lane & 15u;
    unsigned kh   = lane >> 4;                 // 0 or 1
    const _Float16* row = p + (size_t)m * ld + kh * 8;
    v8h lo = *(const v8h*)row;                 // K = kh*8 .. kh*8+7
    v16h out;
    if (TAIL16) {
#pragma unroll
        for (int i = 0; i < 8; ++i) { out[i] = lo[i]; out[8 + i] = (_Float16)0.0f; }
    } else {
        v8h hi = *(const v8h*)(row + 16);      // K = 16 + kh*8 ..
#pragma unroll
        for (int i = 0; i < 8; ++i) { out[i] = lo[i]; out[8 + i] = hi[i]; }
    }
    return out;
}

// C/D 16x16 f32: VGPR r -> (M = r + 8*(lane>=16), N = lane&15)
__device__ __forceinline__ void store_frag(float* p, int ld, v8f c) {
    unsigned lane = threadIdx.x & 31u;
    unsigned n    = lane & 15u;
    unsigned mb   = (lane >> 4) * 8;
#pragma unroll
    for (int r = 0; r < 8; ++r) p[(size_t)(mb + r) * ld + n] = c[r];
}

__device__ __forceinline__ v8f wmma_f16(v16h a, v16h b, v8f c) {
    return __builtin_amdgcn_wmma_f32_16x16x32_f16(false, a, false, b, (short)0, c,
                                                  false, false);
}

// ---- Kernel 1: (b,n,t) f32 -> (t, b*n) f16 tiled transpose ----------------
__global__ __launch_bounds__(256) void k_transpose(const float* __restrict__ in,
                                                   _Float16* __restrict__ out) {
    __shared__ float tile[32][33];
    int t0 = blockIdx.x * 32;   // time tile
    int r0 = blockIdx.y * 32;   // row tile (rows = b*784+n, 200704 % 32 == 0)
#pragma unroll
    for (int k = 0; k < 4; ++k) {
        int r = r0 + threadIdx.y + k * 8;
        int t = t0 + threadIdx.x;
        tile[threadIdx.y + k * 8][threadIdx.x] =
            (t < TSTEPS) ? __builtin_nontemporal_load(&in[(size_t)r * TSTEPS + t])
                         : 0.0f;
    }
    __syncthreads();
#pragma unroll
    for (int k = 0; k < 4; ++k) {
        int t = t0 + threadIdx.y + k * 8;
        int r = r0 + threadIdx.x;
        if (t < TSTEPS)
            out[(size_t)t * ROWS + r] = (_Float16)tile[threadIdx.x][threadIdx.y + k * 8];
    }
}

// ---- Kernel 2a: wfeT[e][n] = relu(w_input_exc[n][e]) ----------------------
__global__ void k_prep_wfe(const float* __restrict__ w, _Float16* __restrict__ wfeT) {
    int i = blockIdx.x * blockDim.x + threadIdx.x;
    if (i >= N_EXC * N_INPUT) return;
    int e = i / N_INPUT, n = i % N_INPUT;
    wfeT[i] = (_Float16)fmaxf(w[(size_t)n * N_EXC + e], 0.0f);
}

// ---- Kernel 2b: wieT[e][k] = relu(w_inh_exc[k][e])*mask, K padded to 416 --
__global__ void k_prep_wie(const float* __restrict__ w, const float* __restrict__ mask,
                           _Float16* __restrict__ wieT) {
    int i = blockIdx.x * blockDim.x + threadIdx.x;
    if (i >= N_EXC * K_INH_PAD) return;
    int e = i / K_INH_PAD, k = i % K_INH_PAD;
    float v = 0.0f;
    if (k < N_INH)
        v = fmaxf(w[(size_t)k * N_EXC + e], 0.0f) * mask[(size_t)k * N_EXC + e];
    wieT[i] = (_Float16)v;
}

// ---- Kernel 3: feed-forward GEMM, 32x80 output tile per wave --------------
// Tiles: 350 t * 8 b-tiles(32) * 5 e-tiles(80) = 14000 waves = 3500 blocks.
// Per K-step: 2 A frags + 5 B frags (14 b128 loads) feed 10 WMMAs -> 1.4
// loads/WMMA (vs 4.0 with one tile/wave), cutting L2 fragment traffic ~2.9x.
__global__ __launch_bounds__(128) void k_ff_gemm(const _Float16* __restrict__ spikes,
                                                 const _Float16* __restrict__ wfeT,
                                                 float* __restrict__ I_all) {
    int wave   = threadIdx.x >> 5;
    int tileId = blockIdx.x * 4 + wave;
    int t   = tileId / 40;      // 8 * 5 tiles per timestep
    int rem = tileId % 40;
    int bt  = rem / 5;          // batch tile of 32 rows
    int et  = rem % 5;          // exc tile of 80 cols

    const _Float16* A = spikes + ((size_t)t * BATCH + bt * 32) * N_INPUT;
    const _Float16* B = wfeT + (size_t)(et * 80) * N_INPUT;

    v8f c[2][5];
#pragma unroll
    for (int sm = 0; sm < 2; ++sm)
#pragma unroll
        for (int sn = 0; sn < 5; ++sn) c[sm][sn] = (v8f){};

    for (int k0 = 0; k0 < 768; k0 += 32) {               // 784 = 24*32 + 16
        v16h a0 = load_frag<false>(A + k0, N_INPUT);
        v16h a1 = load_frag<false>(A + (size_t)16 * N_INPUT + k0, N_INPUT);
        v16h b[5];
#pragma unroll
        for (int sn = 0; sn < 5; ++sn)
            b[sn] = load_frag<false>(B + (size_t)(sn * 16) * N_INPUT + k0, N_INPUT);
#pragma unroll
        for (int sn = 0; sn < 5; ++sn) {
            c[0][sn] = wmma_f16(a0, b[sn], c[0][sn]);
            c[1][sn] = wmma_f16(a1, b[sn], c[1][sn]);
        }
    }
    {                                                    // K tail of 16
        v16h a0 = load_frag<true>(A + 768, N_INPUT);
        v16h a1 = load_frag<true>(A + (size_t)16 * N_INPUT + 768, N_INPUT);
#pragma unroll
        for (int sn = 0; sn < 5; ++sn) {
            v16h b = load_frag<true>(B + (size_t)(sn * 16) * N_INPUT + 768, N_INPUT);
            c[0][sn] = wmma_f16(a0, b, c[0][sn]);
            c[1][sn] = wmma_f16(a1, b, c[1][sn]);
        }
    }
    float* O = I_all + ((size_t)t * BATCH + bt * 32) * N_EXC + et * 80;
#pragma unroll
    for (int sm = 0; sm < 2; ++sm)
#pragma unroll
        for (int sn = 0; sn < 5; ++sn)
            store_frag(O + (size_t)(sm * 16) * N_EXC + sn * 16, N_EXC, c[sm][sn]);
}

// ---- Kernel 4: persistent scan. 16 WGs x 256 thr; 16 batch rows each ------
__global__ __launch_bounds__(256) void k_scan(const float* __restrict__ I_all,
                                              const _Float16* __restrict__ wieT,
                                              float* __restrict__ out_exc,
                                              float* __restrict__ out_inh) {
    __shared__ _Float16 s_inh[16][K_INH_PAD];   // inhibitory spikes (A matrix)
    __shared__ float    R[16][N_EXC];           // s_inh_prev @ w_ie

    const int tid  = threadIdx.x;
    const int wave = tid >> 5;
    const int b0   = blockIdx.x * 16;

    // Per-thread state: element i = tid + 256*j, j=0..24 (16*400/256 == 25)
    float    ve[25], vi[25], se[25];
    unsigned be[25], bi[25];
#pragma unroll
    for (int j = 0; j < 25; ++j) {
        ve[j] = EXC_VREST; vi[j] = INH_VREST; se[j] = 0.0f; be[j] = 0u; bi[j] = 0u;
    }
    for (int i = tid; i < 16 * K_INH_PAD; i += 256) (&s_inh[0][0])[i] = (_Float16)0.0f;
    __syncthreads();

    for (int t = 0; t < TSTEPS; ++t) {
        // Prefetch next timestep's feed-forward slice (hides L2/HBM latency
        // behind the recurrent WMMA phase). 256 thr x 1 line covers 25.6KB.
        if (t + 1 < TSTEPS) {
            const float* nxt = I_all + ((size_t)(t + 1) * BATCH + b0) * N_EXC;
            __builtin_prefetch(nxt + tid * 25, 0, 1);
        }

        // Phase 1: R = s_inh_prev @ w_ie  (K = 416 = 13 * 32, no tail)
        for (int et = wave; et < 25; et += 8) {
            v8f c = (v8f){};
            const _Float16* B = wieT + (size_t)(et * 16) * K_INH_PAD;
#pragma unroll
            for (int k0 = 0; k0 < K_INH_PAD; k0 += 32) {
                v16h a = load_frag<false>(&s_inh[0][0] + k0, K_INH_PAD);
                v16h b = load_frag<false>(B + k0, K_INH_PAD);
                c = wmma_f16(a, b, c);
            }
            store_frag(&R[0][0] + et * 16, N_EXC, c);
        }
        __syncthreads();

        // Phase 2: LIF updates (each element owned by exactly one thread)
        const float* Iff = I_all + ((size_t)t * BATCH + b0) * N_EXC;
        int bit = t & 31;
#pragma unroll
        for (int j = 0; j < 25; ++j) {
            int i = tid + j * 256;
            int row = i / N_EXC, e = i % N_EXC;
            // inhibitory neuron: driven by previous exc spike (one-to-one)
            float v_i = INH_VREST + INH_DECAY * (vi[j] - INH_VREST) + WEIGHT_EI * se[j];
            float s_i = (v_i >= INH_VTH) ? 1.0f : 0.0f;
            vi[j] = (s_i > 0.0f) ? INH_VRESET : v_i;
            s_inh[row][e] = (_Float16)s_i;
            // excitatory neuron
            float I   = Iff[(size_t)row * N_EXC + e] - R[row][e];
            float v_e = EXC_VREST + EXC_DECAY * (ve[j] - EXC_VREST) + I;
            float s_e = (v_e >= EXC_VTH) ? 1.0f : 0.0f;
            ve[j] = (s_e > 0.0f) ? EXC_VRESET : v_e;
            se[j] = s_e;
            if (s_e > 0.0f) be[j] |= (1u << bit);
            if (s_i > 0.0f) bi[j] |= (1u << bit);
        }
        __syncthreads();

        // Flush spike bitmasks every 32 steps: 128B contiguous NT-store run
        // per thread (write-once raster; keep L2 for reused weights/I_all).
        if ((t & 31) == 31 || t == TSTEPS - 1) {
            int tbase = t & ~31;
            int nb    = t - tbase + 1;
#pragma unroll
            for (int j = 0; j < 25; ++j) {
                int i = tid + j * 256;
                int row = i / N_EXC, e = i % N_EXC;
                size_t off = ((size_t)(b0 + row) * N_EXC + e) * TSTEPS + tbase;
                unsigned mbe = be[j], mbi = bi[j];
                for (int q = 0; q < nb; ++q) {
                    __builtin_nontemporal_store(((mbe >> q) & 1u) ? 1.0f : 0.0f,
                                                &out_exc[off + q]);
                    __builtin_nontemporal_store(((mbi >> q) & 1u) ? 1.0f : 0.0f,
                                                &out_inh[off + q]);
                }
                be[j] = 0u; bi[j] = 0u;
            }
        }
    }
}

// ---------------------------------------------------------------------------
extern "C" void kernel_launch(void* const* d_in, const int* in_sizes, int n_in,
                              void* d_out, int out_size, void* d_ws, size_t ws_size,
                              hipStream_t stream) {
    const float* input_spikes = (const float*)d_in[0]; // (256,784,350)
    const float* w_input_exc  = (const float*)d_in[1]; // (784,400)
    const float* w_inh_exc    = (const float*)d_in[2]; // (400,400)
    const float* inh_exc_mask = (const float*)d_in[3]; // (400,400)
    float* out = (float*)d_out;                        // exc | inh rasters

    // Workspace layout (all offsets 256B aligned):
    char* ws = (char*)d_ws;
    const size_t OFF_SPIKES = 0;                                  // 140,492,800 B
    const size_t OFF_WFE    = OFF_SPIKES + (size_t)TSTEPS * ROWS * 2;
    const size_t OFF_WIE    = OFF_WFE + (size_t)N_EXC * N_INPUT * 2;   // 627,200 B
    const size_t OFF_IALL   = OFF_WIE + (size_t)N_EXC * K_INH_PAD * 2; // 332,800 B
    _Float16* spikes_h = (_Float16*)(ws + OFF_SPIKES);
    _Float16* wfeT     = (_Float16*)(ws + OFF_WFE);
    _Float16* wieT     = (_Float16*)(ws + OFF_WIE);
    float*    I_all    = (float*)(ws + OFF_IALL);                 // 143,360,000 B

    k_transpose<<<dim3((TSTEPS + 31) / 32, ROWS / 32), dim3(32, 8), 0, stream>>>(
        input_spikes, spikes_h);
    k_prep_wfe<<<(N_EXC * N_INPUT + 255) / 256, 256, 0, stream>>>(w_input_exc, wfeT);
    k_prep_wie<<<(N_EXC * K_INH_PAD + 255) / 256, 256, 0, stream>>>(w_inh_exc,
                                                                    inh_exc_mask, wieT);
    k_ff_gemm<<<TSTEPS * 8 * 5 / 4, 128, 0, stream>>>(spikes_h, wfeT, I_all);
    k_scan<<<BATCH / 16, 256, 0, stream>>>(I_all, wieT, out,
                                           out + (size_t)BATCH * N_EXC * TSTEPS);
}
